// DecoderBlock_85100482003058
// MI455X (gfx1250) — compile-verified, matrix operand order checked by reading, not run on hardware
//
#include <hip/hip_runtime.h>
#include <hip/hip_bf16.h>

typedef __bf16 bf16;
typedef __attribute__((ext_vector_type(16))) __bf16 v16bf;
typedef __attribute__((ext_vector_type(8)))  __bf16 bf16x8;   // 16 bytes
typedef __attribute__((ext_vector_type(8)))  float  v8f;
typedef int v4i __attribute__((vector_size(16)));             // int4, 16 bytes

// Async global->LDS staging (gfx1250 GLOBAL_LOAD_ASYNC_TO_LDS_B128, ASYNCcnt),
// used when the toolchain exposes the builtins; otherwise fall back to the
// global_load_b128 + ds_store_b128 staging.
#if defined(__has_builtin)
#  if __has_builtin(__builtin_amdgcn_global_load_async_to_lds_b128) && \
      __has_builtin(__builtin_amdgcn_s_wait_asynccnt)
#    define USE_ASYNC_LDS 1
#  endif
#endif
#ifndef USE_ASYNC_LDS
#  define USE_ASYNC_LDS 0
#endif

#define AS1 __attribute__((address_space(1)))
#define AS3 __attribute__((address_space(3)))

// ---------- helpers ----------
__device__ __forceinline__ bf16 to_bf16(float f) {
    unsigned u = __builtin_bit_cast(unsigned, f);
    unsigned r = u + 0x7FFFu + ((u >> 16) & 1u);          // round-to-nearest-even
    unsigned short hv = (unsigned short)(r >> 16);
    return __builtin_bit_cast(bf16, hv);
}
__device__ __forceinline__ float bf2f(bf16 b) {
    unsigned short hv = __builtin_bit_cast(unsigned short, b);
    unsigned u = ((unsigned)hv) << 16;
    return __builtin_bit_cast(float, u);
}
union Frag16 { v16bf v; bf16x8 h[2]; };

__device__ __forceinline__ v8f wmma_bf16(v16bf a, v16bf b, v8f c) {
    // D = A(16x32 bf16) * B(32x16 bf16) + C(16x16 f32)
    return __builtin_amdgcn_wmma_f32_16x16x32_bf16(false, a, false, b, (short)0, c, false, false);
}
__device__ __forceinline__ v8f zero8() {
    v8f z = {0.f,0.f,0.f,0.f,0.f,0.f,0.f,0.f};
    return z;
}

// ---------- 1. ConvPosEnc: depthwise 3x3 + bias + residual ----------
__global__ __launch_bounds__(256) void cpe_kernel(const float* __restrict__ x,
                                                  const float* __restrict__ w,
                                                  const float* __restrict__ bconv,
                                                  float* __restrict__ x0,
                                                  int Hh, int Ww) {
    const int Ntok = Hh * Ww;
    const int n = blockIdx.x % Ntok;
    const int b = blockIdx.x / Ntok;
    const int y = n / Ww, xc = n % Ww;
    const size_t baseB = (size_t)b * Ntok * 512;
    for (int c = threadIdx.x; c < 512; c += 256) {
        float acc = bconv[c];
        #pragma unroll
        for (int ky = 0; ky < 3; ++ky) {
            int yy = y + ky - 1; if (yy < 0 || yy >= Hh) continue;
            #pragma unroll
            for (int kx = 0; kx < 3; ++kx) {
                int xx = xc + kx - 1; if (xx < 0 || xx >= Ww) continue;
                acc += x[baseB + (size_t)(yy * Ww + xx) * 512 + c] * w[c * 9 + ky * 3 + kx];
            }
        }
        x0[baseB + (size_t)n * 512 + c] = acc + x[baseB + (size_t)n * 512 + c];
    }
}

// ---------- 2. LayerNorm (C=512) + cast to bf16 ----------
__global__ __launch_bounds__(256) void ln_cast_kernel(const float* __restrict__ src,
                                                      const float* __restrict__ g,
                                                      const float* __restrict__ bta,
                                                      bf16* __restrict__ dst) {
    const size_t row = blockIdx.x;
    const float* p = src + row * 512;
    const int tid = threadIdx.x;
    float a = p[tid], c = p[tid + 256];
    __shared__ float red[256];
    red[tid] = a + c; __syncthreads();
    for (int s = 128; s > 0; s >>= 1) { if (tid < s) red[tid] += red[tid + s]; __syncthreads(); }
    float mean = red[0] * (1.0f / 512.0f);
    __syncthreads();
    float d0 = a - mean, d1 = c - mean;
    red[tid] = d0 * d0 + d1 * d1; __syncthreads();
    for (int s = 128; s > 0; s >>= 1) { if (tid < s) red[tid] += red[tid + s]; __syncthreads(); }
    float rstd = rsqrtf(red[0] * (1.0f / 512.0f) + 1e-6f);
    dst[row * 512 + tid]       = to_bf16(d0 * rstd * g[tid]       + bta[tid]);
    dst[row * 512 + tid + 256] = to_bf16(d1 * rstd * g[tid + 256] + bta[tid + 256]);
}

// ---------- 3. weight f32 [K,N] -> bf16 transposed [N,K] ----------
__global__ __launch_bounds__(256) void wconv_kernel(const float* __restrict__ w,
                                                    bf16* __restrict__ wt, int K, int Nd) {
    size_t i = (size_t)blockIdx.x * 256 + threadIdx.x;
    if (i >= (size_t)K * Nd) return;
    int k = (int)(i / Nd), n = (int)(i % Nd);
    wt[(size_t)n * K + k] = to_bf16(w[i]);
}

// ---------- 4. generic bf16 WMMA GEMM: C = act(A @ Bt^T + bias) + res ----------
// A: [M,K] bf16 row-major, Bt: [N,K] bf16 (B column-major). Requires M%128==0,N%128==0,K%32==0.
template<int HAS_BIAS, int HAS_RES, int DO_GELU, int OUT_BF16>
__global__ __launch_bounds__(256) void gemm_kernel(const bf16* __restrict__ A,
                                                   const bf16* __restrict__ Bt,
                                                   const float* __restrict__ bias,
                                                   const float* __restrict__ res,
                                                   void* __restrict__ outv,
                                                   int M, int Nd, int K) {
    constexpr int BM = 128, BN = 128, BK = 32, LDSK = 48;   // 96B row stride: 16B aligned
    __shared__ bf16 sA[2][BM][LDSK];
    __shared__ bf16 sB[2][BN][LDSK];
    const int tid  = threadIdx.x;
    const int lane = tid & 31, w = tid >> 5;
    const int l16  = lane & 15, kh = lane >> 4;
    const int wm   = w & 3,  wn = w >> 2;                   // 4x2 wave grid
    const int bm   = blockIdx.y * BM, bn = blockIdx.x * BN;

    v8f acc[2][4];
    #pragma unroll
    for (int i = 0; i < 2; ++i)
        #pragma unroll
        for (int j = 0; j < 4; ++j) acc[i][j] = zero8();

    auto stage = [&](int s, int k0) {
        #pragma unroll
        for (int i = 0; i < 2; ++i) {
            int lin = tid + i * 256;
            int r   = lin >> 2;
            int c8  = (lin & 3) << 3;
#if USE_ASYNC_LDS
            __builtin_amdgcn_global_load_async_to_lds_b128(
                (AS1 v4i*)(A  + (size_t)(bm + r) * K + k0 + c8),
                (AS3 v4i*)&sA[s][r][c8], 0, 0);
            __builtin_amdgcn_global_load_async_to_lds_b128(
                (AS1 v4i*)(Bt + (size_t)(bn + r) * K + k0 + c8),
                (AS3 v4i*)&sB[s][r][c8], 0, 0);
#else
            *(bf16x8*)&sA[s][r][c8] = *(const bf16x8*)(A  + (size_t)(bm + r) * K + k0 + c8);
            *(bf16x8*)&sB[s][r][c8] = *(const bf16x8*)(Bt + (size_t)(bn + r) * K + k0 + c8);
#endif
        }
    };

    const int nk = K / BK;
    stage(0, 0);
#if USE_ASYNC_LDS
    __builtin_amdgcn_s_wait_asynccnt(0);
#endif
    __syncthreads();
    for (int kt = 0; kt < nk; ++kt) {
        const int buf = kt & 1;
        if (kt + 1 < nk) stage(buf ^ 1, (kt + 1) * BK);     // fill other buffer, overlap
        Frag16 fa[2], fb[4];
        #pragma unroll
        for (int i = 0; i < 2; ++i) {
            int row = wm * 32 + i * 16 + l16;
            fa[i].h[0] = *(const bf16x8*)&sA[buf][row][kh * 8];
            fa[i].h[1] = *(const bf16x8*)&sA[buf][row][kh * 8 + 16];
        }
        #pragma unroll
        for (int j = 0; j < 4; ++j) {
            int col = wn * 64 + j * 16 + l16;
            fb[j].h[0] = *(const bf16x8*)&sB[buf][col][kh * 8];
            fb[j].h[1] = *(const bf16x8*)&sB[buf][col][kh * 8 + 16];
        }
        #pragma unroll
        for (int i = 0; i < 2; ++i)
            #pragma unroll
            for (int j = 0; j < 4; ++j)
                acc[i][j] = wmma_bf16(fa[i].v, fb[j].v, acc[i][j]);
#if USE_ASYNC_LDS
        __builtin_amdgcn_s_wait_asynccnt(0);
#endif
        __syncthreads();
    }

    const int rofs = kh * 8;                                // C/D layout: row = r + 8*(lane>=16)
    #pragma unroll
    for (int i = 0; i < 2; ++i)
        #pragma unroll
        for (int j = 0; j < 4; ++j)
            #pragma unroll
            for (int r = 0; r < 8; ++r) {
                int gm = bm + wm * 32 + i * 16 + rofs + r;
                int gn = bn + wn * 64 + j * 16 + l16;
                float v = acc[i][j][r];
                if (HAS_BIAS) v += bias[gn];
                if (DO_GELU)  v = 0.5f * v * (1.0f + erff(v * 0.70710678f));
                if (HAS_RES)  v += res[(size_t)gm * Nd + gn];
                if (OUT_BF16) ((bf16*)outv)[(size_t)gm * Nd + gn] = to_bf16(v);
                else          ((float*)outv)[(size_t)gm * Nd + gn] = v;
            }
}

// ---------- 5. softmax of k over tokens + transposed bf16 copies ----------
// out: ksmt[b,h,ch,n], vt[b,h,ch,n]
__global__ __launch_bounds__(256) void softmax_t_kernel(const bf16* __restrict__ qkv,
                                                        bf16* __restrict__ ksmt,
                                                        bf16* __restrict__ vt, int Ntok) {
    const int g  = blockIdx.x;                // B*8*64
    const int ch = g & 63;
    const int bh = g >> 6;
    const int b  = bh >> 3, hh = bh & 7;
    const bf16* kp = qkv + ((size_t)b * Ntok) * 1536 +  512 + hh * 64 + ch;
    const bf16* vp = qkv + ((size_t)b * Ntok) * 1536 + 1024 + hh * 64 + ch;
    __shared__ float red[256];
    const int tid = threadIdx.x;
    float mx = -1e30f;
    for (int n = tid; n < Ntok; n += 256) mx = fmaxf(mx, bf2f(kp[(size_t)n * 1536]));
    red[tid] = mx; __syncthreads();
    for (int s = 128; s > 0; s >>= 1) { if (tid < s) red[tid] = fmaxf(red[tid], red[tid + s]); __syncthreads(); }
    mx = red[0]; __syncthreads();
    float sum = 0.f;
    for (int n = tid; n < Ntok; n += 256) sum += __expf(bf2f(kp[(size_t)n * 1536]) - mx);
    red[tid] = sum; __syncthreads();
    for (int s = 128; s > 0; s >>= 1) { if (tid < s) red[tid] += red[tid + s]; __syncthreads(); }
    const float inv = 1.0f / red[0];
    bf16* kd = ksmt + ((size_t)bh * 64 + ch) * Ntok;
    bf16* vd = vt   + ((size_t)bh * 64 + ch) * Ntok;
    for (int n = tid; n < Ntok; n += 256) {
        kd[n] = to_bf16(__expf(bf2f(kp[(size_t)n * 1536]) - mx) * inv);
        vd[n] = vp[(size_t)n * 1536];
    }
}

// ---------- 6. kv = k_sm^T @ v per (b,h): 64x64, K=Ntok; output kv^T bf16 ----------
__global__ __launch_bounds__(256) void kv_kernel(const bf16* __restrict__ ksmt,
                                                 const bf16* __restrict__ vt,
                                                 bf16* __restrict__ kvT, int Ntok) {
    const int bh  = blockIdx.x;               // B*8
    const int tid = threadIdx.x, lane = tid & 31, w = tid >> 5;
    const int l16 = lane & 15, kh = lane >> 4;
    const int mt  = w >> 1;                   // ch_k tile 0..3
    const int ntb = (w & 1) * 2;              // ch_v tile base 0 or 2
    const bf16* Ab = ksmt + (size_t)bh * 64 * Ntok;
    const bf16* Bb = vt   + (size_t)bh * 64 * Ntok;
    v8f acc0 = zero8(), acc1 = zero8();
    for (int k0 = 0; k0 < Ntok; k0 += 32) {
        Frag16 a, b0, b1;
        const bf16* pa = Ab + (size_t)(mt * 16 + l16) * Ntok + k0 + kh * 8;
        a.h[0] = *(const bf16x8*)pa;  a.h[1] = *(const bf16x8*)(pa + 16);
        const bf16* p0 = Bb + (size_t)(ntb * 16 + l16) * Ntok + k0 + kh * 8;
        b0.h[0] = *(const bf16x8*)p0; b0.h[1] = *(const bf16x8*)(p0 + 16);
        const bf16* p1 = Bb + (size_t)((ntb + 1) * 16 + l16) * Ntok + k0 + kh * 8;
        b1.h[0] = *(const bf16x8*)p1; b1.h[1] = *(const bf16x8*)(p1 + 16);
        acc0 = wmma_bf16(a.v, b0.v, acc0);
        acc1 = wmma_bf16(a.v, b1.v, acc1);
    }
    const int rofs = kh * 8;
    #pragma unroll
    for (int r = 0; r < 8; ++r) {
        int chk = mt * 16 + rofs + r;
        kvT[(size_t)bh * 4096 + (size_t)(ntb * 16 + l16) * 64 + chk]       = to_bf16(acc0[r]);
        kvT[(size_t)bh * 4096 + (size_t)((ntb + 1) * 16 + l16) * 64 + chk] = to_bf16(acc1[r]);
    }
}

// ---------- 7. ConvRelPosEnc: grouped depthwise 3/5/7 on v ----------
__global__ __launch_bounds__(256) void crpe_conv_kernel(const bf16* __restrict__ qkv,
                                                        const float* __restrict__ w3, const float* __restrict__ b3,
                                                        const float* __restrict__ w5, const float* __restrict__ b5,
                                                        const float* __restrict__ w7, const float* __restrict__ b7,
                                                        bf16* __restrict__ convv, int Hh, int Ww) {
    const int Ntok = Hh * Ww;
    const int n = blockIdx.x % Ntok;
    const int b = blockIdx.x / Ntok;
    const int y = n / Ww, xc = n % Ww;
    for (int c = threadIdx.x; c < 512; c += 256) {
        int ksz; const float* wp; float acc;
        if (c < 128)      { ksz = 3; wp = w3 + c * 9;          acc = b3[c]; }
        else if (c < 320) { ksz = 5; wp = w5 + (c - 128) * 25; acc = b5[c - 128]; }
        else              { ksz = 7; wp = w7 + (c - 320) * 49; acc = b7[c - 320]; }
        const int pad = ksz >> 1;
        for (int ky = 0; ky < ksz; ++ky) {
            int yy = y + ky - pad; if (yy < 0 || yy >= Hh) continue;
            for (int kx = 0; kx < ksz; ++kx) {
                int xx = xc + kx - pad; if (xx < 0 || xx >= Ww) continue;
                acc += bf2f(qkv[((size_t)(b * Ntok + yy * Ww + xx)) * 1536 + 1024 + c]) * wp[ky * ksz + kx];
            }
        }
        convv[((size_t)(b * Ntok + n)) * 512 + c] = to_bf16(acc);
    }
}

// ---------- 8. factor_att = q @ kv (WMMA) fused with crpe combine ----------
// attn_in[b,n,c] = scale*fa + q*conv_v   (bf16, ready for proj GEMM)
__global__ __launch_bounds__(128) void fatt_kernel(const bf16* __restrict__ qkv,
                                                   const bf16* __restrict__ kvT,
                                                   const bf16* __restrict__ convv,
                                                   bf16* __restrict__ attn_in, int Ntok) {
    const int nblk = Ntok / 64;
    const int g  = blockIdx.x;                // B*8*nblk
    const int nb = g % nblk;
    const int bh = g / nblk;
    const int b  = bh >> 3, hh = bh & 7;
    const int tid = threadIdx.x, lane = tid & 31, w = tid >> 5;   // 4 waves
    const int l16 = lane & 15, kh = lane >> 4;
    const int rowBase = nb * 64 + w * 16;
    v8f acc[4] = {zero8(), zero8(), zero8(), zero8()};
    #pragma unroll
    for (int k0 = 0; k0 < 64; k0 += 32) {
        Frag16 a;
        {
            int n = rowBase + l16;
            const bf16* pa = qkv + ((size_t)(b * Ntok + n)) * 1536 + hh * 64 + k0 + kh * 8;
            a.h[0] = *(const bf16x8*)pa;
            a.h[1] = *(const bf16x8*)(pa + 16);
        }
        #pragma unroll
        for (int j = 0; j < 4; ++j) {
            Frag16 fb;
            const bf16* pb = kvT + (size_t)bh * 4096 + (size_t)(j * 16 + l16) * 64 + k0 + kh * 8;
            fb.h[0] = *(const bf16x8*)pb;
            fb.h[1] = *(const bf16x8*)(pb + 16);
            acc[j] = wmma_bf16(a.v, fb.v, acc[j]);
        }
    }
    const float scale = 0.125f;               // 64^-0.5
    const int rofs = kh * 8;
    #pragma unroll
    for (int j = 0; j < 4; ++j)
        #pragma unroll
        for (int r = 0; r < 8; ++r) {
            int n = rowBase + rofs + r;
            int c = hh * 64 + j * 16 + l16;
            size_t tok = (size_t)(b * Ntok + n);
            float qv = bf2f(qkv[tok * 1536 + c]);
            float cv = bf2f(convv[tok * 512 + c]);
            attn_in[tok * 512 + c] = to_bf16(scale * acc[j][r] + qv * cv);
        }
}

// ---------- host-side launcher ----------
extern "C" void kernel_launch(void* const* d_in, const int* in_sizes, int n_in,
                              void* d_out, int out_size, void* d_ws, size_t ws_size,
                              hipStream_t stream) {
    const int C = 512, Hh = 56, Ww = 56, Ntok = Hh * Ww, hid = 2048;
    const int Bb = in_sizes[0] / (Ntok * C);
    const size_t M = (size_t)Bb * Ntok;

    const float* x      = (const float*)d_in[0];
    const float* cpe_w  = (const float*)d_in[3];
    const float* cpe_b  = (const float*)d_in[4];
    const float* ln1_w  = (const float*)d_in[5];
    const float* ln1_b  = (const float*)d_in[6];
    const float* qkv_w  = (const float*)d_in[7];
    const float* proj_w = (const float*)d_in[8];
    const float* proj_b = (const float*)d_in[9];
    const float* w3 = (const float*)d_in[10]; const float* b3 = (const float*)d_in[11];
    const float* w5 = (const float*)d_in[12]; const float* b5 = (const float*)d_in[13];
    const float* w7 = (const float*)d_in[14]; const float* b7 = (const float*)d_in[15];
    const float* ln2_w  = (const float*)d_in[16];
    const float* ln2_b  = (const float*)d_in[17];
    const float* fc1_w  = (const float*)d_in[18];
    const float* fc1_b  = (const float*)d_in[19];
    const float* fc2_w  = (const float*)d_in[20];
    const float* fc2_b  = (const float*)d_in[21];

    // workspace carve-up
    char* ws = (char*)d_ws;
    size_t off = 0;
    auto carve = [&](size_t bytes) { void* p = ws + off; off += (bytes + 255) & ~(size_t)255; return p; };
    float* x0      = (float*)carve(M * C * 4);
    bf16*  qkv     = (bf16*) carve(M * 3 * C * 2);
    bf16*  cur     = (bf16*) carve(M * C * 2);      // LN out, later reused as attn_in
    bf16*  wqkvt   = (bf16*) carve((size_t)3 * C * C * 2);
    bf16*  wprojt  = (bf16*) carve((size_t)C * C * 2);
    bf16*  wfc1t   = (bf16*) carve((size_t)hid * C * 2);
    bf16*  wfc2t   = (bf16*) carve((size_t)C * hid * 2);
    // transient region (attention scratch), aliased later by FFN hidden
    char*  trans   = ws + off;
    bf16*  ksmt    = (bf16*)(trans);
    bf16*  vt      = (bf16*)(trans + M * C * 2);
    bf16*  kvT     = (bf16*)(trans + 2 * M * C * 2);
    bf16*  convv   = (bf16*)(trans + 2 * M * C * 2 + (size_t)Bb * 8 * 4096 * 2);
    bf16*  hdn     = (bf16*)(trans);                // FFN hidden aliases attention scratch

    const unsigned gy = (unsigned)(M / 128);

    // weight conversions (transposed bf16)
    wconv_kernel<<<(unsigned)((512 * 1536 + 255) / 256), 256, 0, stream>>>(qkv_w,  wqkvt, 512, 1536);
    wconv_kernel<<<(unsigned)((512 * 512  + 255) / 256), 256, 0, stream>>>(proj_w, wprojt, 512, 512);
    wconv_kernel<<<(unsigned)((512 * 2048 + 255) / 256), 256, 0, stream>>>(fc1_w,  wfc1t, 512, 2048);
    wconv_kernel<<<(unsigned)((2048 * 512 + 255) / 256), 256, 0, stream>>>(fc2_w,  wfc2t, 2048, 512);

    // 1. CPE + residual
    cpe_kernel<<<(unsigned)(Bb * Ntok), 256, 0, stream>>>(x, cpe_w, cpe_b, x0, Hh, Ww);
    // 2. LN1 -> bf16
    ln_cast_kernel<<<(unsigned)M, 256, 0, stream>>>(x0, ln1_w, ln1_b, cur);
    // 3. qkv GEMM (bf16 out)
    gemm_kernel<0,0,0,1><<<dim3(1536 / 128, gy), 256, 0, stream>>>(cur, wqkvt, nullptr, nullptr, qkv, (int)M, 1536, 512);
    // 4. softmax(k) over tokens + transposed k_sm / v
    softmax_t_kernel<<<(unsigned)(Bb * 8 * 64), 256, 0, stream>>>(qkv, ksmt, vt, Ntok);
    // 5. kv = k_sm^T v  (per b,h)
    kv_kernel<<<(unsigned)(Bb * 8), 256, 0, stream>>>(ksmt, vt, kvT, Ntok);
    // 6. conv rel pos enc on v
    crpe_conv_kernel<<<(unsigned)(Bb * Ntok), 256, 0, stream>>>(qkv, w3, b3, w5, b5, w7, b7, convv, Hh, Ww);
    // 7. factor_att + crpe combine -> attn_in (reuse `cur`)
    fatt_kernel<<<(unsigned)(Bb * 8 * (Ntok / 64)), 128, 0, stream>>>(qkv, kvT, convv, cur, Ntok);
    // 8. proj GEMM + bias + residual(x0) -> x0 (f32)
    gemm_kernel<1,1,0,0><<<dim3(512 / 128, gy), 256, 0, stream>>>(cur, wprojt, proj_b, x0, x0, (int)M, 512, 512);
    // 9. LN2 -> bf16
    ln_cast_kernel<<<(unsigned)M, 256, 0, stream>>>(x0, ln2_w, ln2_b, cur);
    // 10. fc1 GEMM + bias + GELU -> hdn (bf16)
    gemm_kernel<1,0,1,1><<<dim3(2048 / 128, gy), 256, 0, stream>>>(cur, wfc1t, fc1_b, nullptr, hdn, (int)M, 2048, 512);
    // 11. fc2 GEMM + bias + residual(x0) -> d_out (f32)
    gemm_kernel<1,1,0,0><<<dim3(512 / 128, gy), 256, 0, stream>>>(hdn, wfc2t, fc2_b, x0, d_out, (int)M, 512, 2048);
}
